// MAB_25357486916034
// MI455X (gfx1250) — matrix operand order repaired and checked by reading, not compiled, not run on hardware
//
#include <hip/hip_runtime.h>

// Problem constants (from reference): B=4, SQ=SK=1024, D=1024, H=16, DH=64
#define B_SZ   4
#define SEQ_Q  1024
#define SEQ_K  1024
#define DMODEL 1024
#define NHEAD  16
#define DHEAD  64
#define MROWS  (B_SZ * SEQ_Q)
#define APITCH 1032   // LDS score row pitch in floats (1024 + 8 pad, bank skew)
#define BPITCH 40     // LDS B-tile row pitch in bf16 elems (32 data + 8 skew)

typedef __bf16 bf16_t;
typedef bf16_t v16bf __attribute__((ext_vector_type(16)));
typedef float  v8f   __attribute__((ext_vector_type(8)));

union Frag16 { v16bf v; uint4 q[2]; unsigned short u[16]; };

// fp32 -> bf16, round-to-nearest-even, Inf/NaN passthrough
__device__ __forceinline__ unsigned short f2bf(float f) {
  union { float f; unsigned int u; } x; x.f = f;
  unsigned int r = x.u + 0x7FFFu + ((x.u >> 16) & 1u);
  if (((x.u >> 23) & 0xFFu) == 0xFFu) r = x.u;
  return (unsigned short)(r >> 16);
}

// B-fragment (K x 16, 16-bit): lane n = column, elements j = B[(lane>>4)*16+j][n].
__device__ __forceinline__ v16bf load16(const unsigned short* p) {
  Frag16 f;
  f.q[0] = *(const uint4*)(p);
  f.q[1] = *(const uint4*)(p + 8);
  return f.v;
}

// A-fragment (16 x 32, 16-bit): lane m = lane&15; elems 0..15 = K[kk + (lane>>4)*16 ..]
// (contiguous k order, matching the B fragments)
__device__ __forceinline__ v16bf loadA(const unsigned short* row, int kk, int lane) {
  const int half8 = (lane >> 4) * 16;
  Frag16 f;
  f.q[0] = *(const uint4*)(row + kk + half8);
  f.q[1] = *(const uint4*)(row + kk + half8 + 8);
  return f.v;
}

// CDNA5 async copy: 16 bytes global -> LDS, tracked by ASYNCcnt.
__device__ __forceinline__ void async_b128(unsigned lds_off, unsigned long long gaddr) {
  asm volatile("global_load_async_to_lds_b128 %0, %1, off"
               :: "v"(lds_off), "v"(gaddr)
               : "memory");
}
__device__ __forceinline__ void wait_async2() {
  asm volatile("s_wait_asynccnt 0x2" ::: "memory");
}
__device__ __forceinline__ void wait_async0() {
  asm volatile("s_wait_asynccnt 0x0" ::: "memory");
}

// 4 B-fragments from an LDS tile + 4 WMMAs (one A fragment) — attention P@V.
__device__ __forceinline__ void consume_tile(const unsigned short* bt, int boff,
                                             v16bf a, v8f acc[4]) {
  v16bf b0 = load16(bt + boff);
  v16bf b1 = load16(bt + boff + 16 * BPITCH);
  v16bf b2 = load16(bt + boff + 32 * BPITCH);
  v16bf b3 = load16(bt + boff + 48 * BPITCH);
  acc[0] = __builtin_amdgcn_wmma_f32_16x16x32_bf16(false, a, false, b0, (short)0, acc[0], false, false);
  acc[1] = __builtin_amdgcn_wmma_f32_16x16x32_bf16(false, a, false, b1, (short)0, acc[1], false, false);
  acc[2] = __builtin_amdgcn_wmma_f32_16x16x32_bf16(false, a, false, b2, (short)0, acc[2], false, false);
  acc[3] = __builtin_amdgcn_wmma_f32_16x16x32_bf16(false, a, false, b3, (short)0, acc[3], false, false);
}

// 4 B-fragments + 8 WMMAs (two A fragments, 32x64 wave tile) — GEMM.
__device__ __forceinline__ void consume_tile2(const unsigned short* bt, int boff,
                                              v16bf a0, v16bf a1, v8f acc[8]) {
  v16bf b0 = load16(bt + boff);
  v16bf b1 = load16(bt + boff + 16 * BPITCH);
  v16bf b2 = load16(bt + boff + 32 * BPITCH);
  v16bf b3 = load16(bt + boff + 48 * BPITCH);
  acc[0] = __builtin_amdgcn_wmma_f32_16x16x32_bf16(false, a0, false, b0, (short)0, acc[0], false, false);
  acc[1] = __builtin_amdgcn_wmma_f32_16x16x32_bf16(false, a0, false, b1, (short)0, acc[1], false, false);
  acc[2] = __builtin_amdgcn_wmma_f32_16x16x32_bf16(false, a0, false, b2, (short)0, acc[2], false, false);
  acc[3] = __builtin_amdgcn_wmma_f32_16x16x32_bf16(false, a0, false, b3, (short)0, acc[3], false, false);
  acc[4] = __builtin_amdgcn_wmma_f32_16x16x32_bf16(false, a1, false, b0, (short)0, acc[4], false, false);
  acc[5] = __builtin_amdgcn_wmma_f32_16x16x32_bf16(false, a1, false, b1, (short)0, acc[5], false, false);
  acc[6] = __builtin_amdgcn_wmma_f32_16x16x32_bf16(false, a1, false, b2, (short)0, acc[6], false, false);
  acc[7] = __builtin_amdgcn_wmma_f32_16x16x32_bf16(false, a1, false, b3, (short)0, acc[7], false, false);
}

// ---------------------------------------------------------------------------
// fp32 -> bf16 elementwise convert
// ---------------------------------------------------------------------------
__global__ void cvt_bf16_kernel(const float* __restrict__ x,
                                unsigned short* __restrict__ y, size_t n) {
  size_t i = (size_t)blockIdx.x * blockDim.x + threadIdx.x;
  const size_t st = (size_t)gridDim.x * blockDim.x;
  for (; i < n; i += st) y[i] = f2bf(x[i]);
}

// ---------------------------------------------------------------------------
// WMMA GEMM:  out[m,n] = epilogue( sum_k A[m,k]*W[n,k] + bias[n] )
// A: bf16 [M,K] row-major; W: bf16 [N,K] row-major (torch Linear x@W^T).
// 128 threads = 4 waves; block tile 128(M) x 64(N); wave tile 32 x 64
// (2 A-frags x 4 N-subtiles = 8 accumulators/wave).
// 4 KB B tile (shared by all waves) double-buffered in LDS via async copies;
// A fragments prefetched one step ahead; branchless steady loop + final step.
// ---------------------------------------------------------------------------
__global__ __launch_bounds__(128)
void gemm_nt_kernel(const unsigned short* __restrict__ A,
                    const unsigned short* __restrict__ W,
                    const float* __restrict__ bias,
                    int M, int N, int K,
                    unsigned short* __restrict__ out_bf,
                    float* __restrict__ out_f32,
                    const float* __restrict__ res,
                    unsigned short* __restrict__ out_vt,
                    int S, int DH) {
  __shared__ alignas(16) unsigned short btile[2][64 * BPITCH];

  const int lane = threadIdx.x & 31;
  const int wv   = threadIdx.x >> 5;
  const int n0   = blockIdx.x * 64;
  const int m0   = blockIdx.y * 128 + wv * 32;

  const unsigned short* arow0 = A + (size_t)(m0 + (lane & 15)) * K;
  const unsigned short* arow1 = A + (size_t)(m0 + 16 + (lane & 15)) * K;
  const int boff = (lane & 15) * BPITCH + (lane >> 4) * 16;

  // async-copy mapping: thread t copies 16 elems (32 B) of the 64x32 B tile
  const int trow  = threadIdx.x >> 1;
  const int thalf = threadIdx.x & 1;
  const unsigned short* gsrc = W + (size_t)(n0 + trow) * K + thalf * 16;
  const unsigned lds0 = (unsigned)(unsigned long long)&btile[0][0];
  const unsigned lds1 = (unsigned)(unsigned long long)&btile[1][0];
  const unsigned dsto = (unsigned)(trow * BPITCH + thalf * 16) * 2;  // bytes

  // prologue: issue tile 0, prefetch A frags 0
  async_b128(lds0 + dsto,      (unsigned long long)(size_t)(gsrc));
  async_b128(lds0 + dsto + 16, (unsigned long long)(size_t)(gsrc + 8));
  v16bf a_cur0 = loadA(arow0, 0, lane);
  v16bf a_cur1 = loadA(arow1, 0, lane);

  v8f acc[8] = {};
  const int NS = K / 32;
  for (int s = 0; s < NS - 1; ++s) {   // steady state: always prefetch s+1
    const unsigned short* bt = (s & 1) ? &btile[1][0] : &btile[0][0];
    const unsigned nxt = (s & 1) ? lds0 : lds1;
    const unsigned short* gs = gsrc + (size_t)(s + 1) * 32;
    async_b128(nxt + dsto,      (unsigned long long)(size_t)(gs));
    async_b128(nxt + dsto + 16, (unsigned long long)(size_t)(gs + 8));
    v16bf a_next0 = loadA(arow0, (s + 1) * 32, lane);
    v16bf a_next1 = loadA(arow1, (s + 1) * 32, lane);
    wait_async2();   // current tile complete; next still in flight
    __syncthreads();
    consume_tile2(bt, boff, a_cur0, a_cur1, acc);
    __syncthreads(); // all waves done reading before buffer is overwritten
    a_cur0 = a_next0;
    a_cur1 = a_next1;
  }
  wait_async0();     // final tile
  __syncthreads();
  consume_tile2(((NS - 1) & 1) ? &btile[1][0] : &btile[0][0], boff,
                a_cur0, a_cur1, acc);

  const int nl = lane & 15;
  const int mb = (lane >> 4) * 8;
#pragma unroll
  for (int g = 0; g < 2; ++g) {        // row groups m0+0..15, m0+16..31
#pragma unroll
    for (int t = 0; t < 4; ++t) {
      const int n = n0 + t * 16 + nl;
      const float bv = bias[n];
      const v8f a = acc[g * 4 + t];
#pragma unroll
      for (int r = 0; r < 8; ++r) {
        const int m = m0 + g * 16 + mb + r;
        float y = a[r] + bv;
        if (res) y = res[(size_t)m * N + n] + fmaxf(y, 0.f);
        if (out_f32) out_f32[(size_t)m * N + n] = y;
        if (out_bf)  out_bf[(size_t)m * N + n] = f2bf(y);
        if (out_vt) {  // store as Vt[b, h, d, s]
          const int b = m / S, sdx = m - b * S;
          const int h = n / DH, d = n - h * DH;
          out_vt[(((size_t)b * (N / DH) + h) * DH + d) * S + sdx] = f2bf(y);
        }
      }
    }
  }
}

// ---------------------------------------------------------------------------
// Fused attention per (q-tile of 64 rows, batch*head):
//   scores = (Qh Kh^T)/sqrt(D), mask -> -inf, exact softmax in LDS,
//   O = Qh_f32 + (P V) (per-head residual).
// 4 waves; each wave owns 16 q-rows. V tiles for P@V are identical across
// waves -> async-staged + double-buffered in LDS like the GEMM B tiles.
// ---------------------------------------------------------------------------
__global__ __launch_bounds__(128)
void attn_kernel(const unsigned short* __restrict__ Qh,   // bf16 [B,SQ,D]
                 const unsigned short* __restrict__ Kh,   // bf16 [B,SK,D]
                 const unsigned short* __restrict__ Vt,   // bf16 [B,H,DH,SK]
                 const float* __restrict__ Qf,            // f32  [B,SQ,D]
                 const int* __restrict__ mask,            // [B,SK]
                 float* __restrict__ Out) {               // f32  [B,SQ,D]
  extern __shared__ float sc[];        // 64 x APITCH scores (f32)
  __shared__ float inv_l[64];
  __shared__ alignas(16) unsigned short vbuf[2][64 * BPITCH];

  const int lane = threadIdx.x & 31;
  const int wv   = threadIdx.x >> 5;
  const int q0   = blockIdx.x * 64;
  const int bh   = blockIdx.y;
  const int b    = bh / NHEAD, h = bh % NHEAD;
  const int mb   = (lane >> 4) * 8;

  const unsigned short* qptr =
      Qh + (size_t)(b * SEQ_Q + q0 + wv * 16 + (lane & 15)) * DMODEL + h * DHEAD;
  const v16bf aq0 = loadA(qptr, 0, lane);
  const v16bf aq1 = loadA(qptr, 32, lane);
  const float scale = 0.03125f;  // 1/sqrt(D), D=1024 (module scales by dim_V!)

  // ---- Phase 1: scores -> LDS ----
  for (int kt = 0; kt < SEQ_K / 16; ++kt) {
    const int key = kt * 16 + (lane & 15);
    const unsigned short* kp =
        Kh + (size_t)(b * SEQ_K + key) * DMODEL + h * DHEAD + (lane >> 4) * 16;
    v16bf bk0 = load16(kp);        // d = 0..31 chunk
    v16bf bk1 = load16(kp + 32);   // d = 32..63 chunk
    v8f c = {};
    c = __builtin_amdgcn_wmma_f32_16x16x32_bf16(false, aq0, false, bk0, (short)0, c, false, false);
    c = __builtin_amdgcn_wmma_f32_16x16x32_bf16(false, aq1, false, bk1, (short)0, c, false, false);
    const int mk = mask[b * SEQ_K + key];
#pragma unroll
    for (int r = 0; r < 8; ++r) {
      const float s = mk ? -__builtin_inff() : c[r] * scale;
      sc[(wv * 16 + mb + r) * APITCH + key] = s;
    }
  }

  // ---- Phase 2: exact row softmax (all-masked rows -> 0, per nan_to_num) ----
  for (int r = 0; r < 16; ++r) {
    float* prow = sc + (wv * 16 + r) * APITCH;
    float mx = -3.4e38f;
    for (int c = lane; c < SEQ_K; c += 32) mx = fmaxf(mx, prow[c]);
#pragma unroll
    for (int off = 16; off; off >>= 1) mx = fmaxf(mx, __shfl_xor(mx, off, 32));
    float sum = 0.f;
    for (int c = lane; c < SEQ_K; c += 32) {
      const float sv = prow[c];
      const float p = (sv > -1e37f) ? __expf(sv - mx) : 0.f;
      prow[c] = p;
      sum += p;
    }
#pragma unroll
    for (int off = 16; off; off >>= 1) sum += __shfl_xor(sum, off, 32);
    if (lane == 0) inv_l[wv * 16 + r] = (sum > 0.f) ? (1.f / sum) : 0.f;
  }

  // ---- Phase 3: P @ V with async-staged, double-buffered V tiles ----
  const int trow  = threadIdx.x >> 1;
  const int thalf = threadIdx.x & 1;
  const unsigned short* vsrc =
      Vt + ((size_t)bh * DHEAD + trow) * SEQ_K + thalf * 16;
  const unsigned vlds0 = (unsigned)(unsigned long long)&vbuf[0][0];
  const unsigned vlds1 = (unsigned)(unsigned long long)&vbuf[1][0];
  const unsigned vdsto = (unsigned)(trow * BPITCH + thalf * 16) * 2;

  async_b128(vlds0 + vdsto,      (unsigned long long)(size_t)(vsrc));
  async_b128(vlds0 + vdsto + 16, (unsigned long long)(size_t)(vsrc + 8));

  v8f o[4] = {};
  const int prow_i = wv * 16 + (lane & 15);
  const int half8  = (lane >> 4) * 16;
  const int boff   = (lane & 15) * BPITCH + (lane >> 4) * 16;
  const int NS = SEQ_K / 32;
  for (int s = 0; s < NS - 1; ++s) {
    const unsigned short* bt = (s & 1) ? &vbuf[1][0] : &vbuf[0][0];
    const unsigned nxt = (s & 1) ? vlds0 : vlds1;
    const unsigned short* gs = vsrc + (size_t)(s + 1) * 32;
    async_b128(nxt + vdsto,      (unsigned long long)(size_t)(gs));
    async_b128(nxt + vdsto + 16, (unsigned long long)(size_t)(gs + 8));
    wait_async2();
    __syncthreads();
    Frag16 fa;
    const int kk = s * 32;
#pragma unroll
    for (int i = 0; i < 8; ++i) {
      fa.u[i]     = f2bf(sc[prow_i * APITCH + kk + half8 + i]);
      fa.u[i + 8] = f2bf(sc[prow_i * APITCH + kk + half8 + 8 + i]);
    }
    consume_tile(bt, boff, fa.v, o);
    __syncthreads();
  }
  wait_async0();
  __syncthreads();
  {
    Frag16 fa;
    const int kk = (NS - 1) * 32;
#pragma unroll
    for (int i = 0; i < 8; ++i) {
      fa.u[i]     = f2bf(sc[prow_i * APITCH + kk + half8 + i]);
      fa.u[i + 8] = f2bf(sc[prow_i * APITCH + kk + half8 + 8 + i]);
    }
    consume_tile(((NS - 1) & 1) ? &vbuf[1][0] : &vbuf[0][0], boff, fa.v, o);
  }

  // ---- Epilogue: O = Qh + P V (unnormalized acc * 1/l) ----
#pragma unroll
  for (int t = 0; t < 4; ++t) {
    const int d = t * 16 + (lane & 15);
#pragma unroll
    for (int r = 0; r < 8; ++r) {
      const int mloc = mb + r;
      const float il = inv_l[wv * 16 + mloc];
      const size_t idx =
          (size_t)(b * SEQ_Q + q0 + wv * 16 + mloc) * DMODEL + h * DHEAD + d;
      Out[idx] = Qf[idx] + o[t][r] * il;
    }
  }
}

// ---------------------------------------------------------------------------
// LayerNorm over D=1024; one 256-thread block per row. Optional bf16 copy.
// ---------------------------------------------------------------------------
__global__ __launch_bounds__(256)
void layernorm_kernel(const float* __restrict__ X, const float* __restrict__ g,
                      const float* __restrict__ be, float* __restrict__ Yf,
                      unsigned short* __restrict__ Ybf) {
  __shared__ float rs[8], rq[8];
  const int row = blockIdx.x;
  const float* x = X + (size_t)row * DMODEL;
  float s = 0.f, ss = 0.f;
  for (int i = threadIdx.x; i < DMODEL; i += 256) {
    const float v = x[i];
    s += v; ss += v * v;
  }
#pragma unroll
  for (int off = 16; off; off >>= 1) {
    s  += __shfl_xor(s, off, 32);
    ss += __shfl_xor(ss, off, 32);
  }
  if ((threadIdx.x & 31) == 0) { rs[threadIdx.x >> 5] = s; rq[threadIdx.x >> 5] = ss; }
  __syncthreads();
  float S = 0.f, SS = 0.f;
#pragma unroll
  for (int i = 0; i < 8; ++i) { S += rs[i]; SS += rq[i]; }
  const float mu  = S * (1.f / DMODEL);
  const float var = SS * (1.f / DMODEL) - mu * mu;
  const float inv = rsqrtf(var + 1e-5f);
  for (int i = threadIdx.x; i < DMODEL; i += 256) {
    const float y = (x[i] - mu) * inv * g[i] + be[i];
    Yf[(size_t)row * DMODEL + i] = y;
    if (Ybf) Ybf[(size_t)row * DMODEL + i] = f2bf(y);
  }
}

// ---------------------------------------------------------------------------
extern "C" void kernel_launch(void* const* d_in, const int* in_sizes, int n_in,
                              void* d_out, int out_size, void* d_ws, size_t ws_size,
                              hipStream_t stream) {
  (void)in_sizes; (void)n_in; (void)out_size; (void)ws_size;
  const float* Q     = (const float*)d_in[0];
  const float* K     = (const float*)d_in[1];
  const int*   mask  = (const int*)d_in[2];
  const float* Wq    = (const float*)d_in[3];
  const float* bq    = (const float*)d_in[4];
  const float* Wk    = (const float*)d_in[5];
  const float* bk    = (const float*)d_in[6];
  const float* Wv    = (const float*)d_in[7];
  const float* bv    = (const float*)d_in[8];
  const float* Wo    = (const float*)d_in[9];
  const float* bo    = (const float*)d_in[10];
  const float* g0    = (const float*)d_in[11];
  const float* beta0 = (const float*)d_in[12];
  const float* g1    = (const float*)d_in[13];
  const float* beta1 = (const float*)d_in[14];
  float* Out = (float*)d_out;

  const size_t MB = 1ull << 20;
  char* ws = (char*)d_ws;
  unsigned short* Qbf   = (unsigned short*)(ws + 0 * MB);   // 8 MB
  unsigned short* Kbf   = (unsigned short*)(ws + 8 * MB);   // 8 MB
  unsigned short* Wqbf  = (unsigned short*)(ws + 16 * MB);  // 2 MB
  unsigned short* Wkbf  = (unsigned short*)(ws + 18 * MB);  // 2 MB
  unsigned short* Wvbf  = (unsigned short*)(ws + 20 * MB);  // 2 MB
  unsigned short* Wobf  = (unsigned short*)(ws + 22 * MB);  // 2 MB
  unsigned short* Qh_bf = (unsigned short*)(ws + 24 * MB);  // 8 MB
  unsigned short* Kh_bf = (unsigned short*)(ws + 32 * MB);  // 8 MB
  unsigned short* Vt_bf = (unsigned short*)(ws + 40 * MB);  // 8 MB
  float* Qh_f = (float*)(ws + 48 * MB);                     // 16 MB
  float* O1   = (float*)(ws + 64 * MB);                     // 16 MB
  // lifetime-based aliasing:
  float* OLn_f = Qh_f;            // Qh_f32 dead after attention
  unsigned short* OLn_bf = Qbf;   // Qbf dead after projections
  float* O2 = O1;                 // O1 dead after LayerNorm0

  const size_t nQK = (size_t)MROWS * DMODEL;   // 4M
  const size_t nW  = (size_t)DMODEL * DMODEL;  // 1M

  // 1) fp32 -> bf16 converts
  cvt_bf16_kernel<<<2048, 256, 0, stream>>>(Q, Qbf, nQK);
  cvt_bf16_kernel<<<2048, 256, 0, stream>>>(K, Kbf, nQK);
  cvt_bf16_kernel<<<1024, 256, 0, stream>>>(Wq, Wqbf, nW);
  cvt_bf16_kernel<<<1024, 256, 0, stream>>>(Wk, Wkbf, nW);
  cvt_bf16_kernel<<<1024, 256, 0, stream>>>(Wv, Wvbf, nW);
  cvt_bf16_kernel<<<1024, 256, 0, stream>>>(Wo, Wobf, nW);

  // 2) projections (block tile 128x64)
  dim3 ggrid(DMODEL / 64, MROWS / 128);
  dim3 gblk(128);
  gemm_nt_kernel<<<ggrid, gblk, 0, stream>>>(Qbf, Wqbf, bq, MROWS, DMODEL, DMODEL,
                                             Qh_bf, Qh_f, nullptr, nullptr, SEQ_Q, DHEAD);
  gemm_nt_kernel<<<ggrid, gblk, 0, stream>>>(Kbf, Wkbf, bk, MROWS, DMODEL, DMODEL,
                                             Kh_bf, nullptr, nullptr, nullptr, SEQ_K, DHEAD);
  gemm_nt_kernel<<<ggrid, gblk, 0, stream>>>(Kbf, Wvbf, bv, MROWS, DMODEL, DMODEL,
                                             nullptr, nullptr, nullptr, Vt_bf, SEQ_K, DHEAD);

  // 3) fused masked attention + per-head residual
  dim3 agrid(SEQ_Q / 64, B_SZ * NHEAD);
  const size_t attn_lds = (size_t)64 * APITCH * sizeof(float);  // 264192 B dynamic
  attn_kernel<<<agrid, dim3(128), attn_lds, stream>>>(Qh_bf, Kh_bf, Vt_bf, Qh_f, mask, O1);

  // 4) LayerNorm0 (f32 + bf16 copy for next GEMM)
  layernorm_kernel<<<MROWS, 256, 0, stream>>>(O1, g0, beta0, OLn_f, OLn_bf);

  // 5) FFN: O2 = OLn + relu(OLn @ Wo^T + bo)
  gemm_nt_kernel<<<ggrid, gblk, 0, stream>>>(OLn_bf, Wobf, bo, MROWS, DMODEL, DMODEL,
                                             nullptr, O2, OLn_f, nullptr, SEQ_Q, DHEAD);

  // 6) LayerNorm1 -> final output
  layernorm_kernel<<<MROWS, 256, 0, stream>>>(O2, g1, beta1, Out, nullptr);
}
